// Tagger4_58136677318791
// MI455X (gfx1250) — matrix operand order repaired and checked by reading, not compiled, not run on hardware
//
#include <hip/hip_runtime.h>
#include <hip/hip_bf16.h>
#include <math.h>

// ---------- sizes ----------
#define BB 16384
#define WW 5
#define LL 20
#define EE 50
#define VV 100000
#define CC 128
#define HH 4096
#define OO 50

typedef _Float16 f16x8 __attribute__((ext_vector_type(8)));
typedef _Float16 v16h  __attribute__((ext_vector_type(16)));
typedef float    v8f   __attribute__((ext_vector_type(8)));

__device__ __forceinline__ v16h load_frag(const _Float16* lo, const _Float16* hi) {
    f16x8 a = *(const f16x8*)lo;
    f16x8 b = *(const f16x8*)hi;
    v16h r;
#pragma unroll
    for (int i = 0; i < 8; ++i) { r[i] = a[i]; r[i + 8] = b[i]; }
    return r;
}

__device__ __forceinline__ v8f wmma16(v16h a, v16h b, v8f c) {
    return __builtin_amdgcn_wmma_f32_16x16x32_f16(
        /*neg_a=*/false, a, /*neg_b=*/false, b,
        /*c_mod=*/(short)0, c, /*reuse_a=*/false, /*reuse_b=*/false);
}

// ---------- prep: fp32 weights -> padded f16, WMMA-friendly layouts ----------
__global__ void k_prep_fc1(const float* __restrict__ w, _Float16* __restrict__ dst) {
    int i = blockIdx.x * 256 + threadIdx.x;          // 4096*256
    int r = i >> 8, c = i & 255;
    dst[i] = (_Float16)(c < 250 ? w[r * 250 + c] : 0.0f);
}
__global__ void k_prep_fc2(const float* __restrict__ w, _Float16* __restrict__ dst) {
    int i = blockIdx.x * 256 + threadIdx.x;          // 64*4096
    int r = i >> 12, c = i & 4095;
    dst[i] = (_Float16)(r < OO ? w[r * HH + c] : 0.0f);
}
__global__ void k_prep_char(const float* __restrict__ e, _Float16* __restrict__ dst) {
    int i = blockIdx.x * 256 + threadIdx.x;          // 128*64
    int r = i >> 6, c = i & 63;
    dst[i] = (_Float16)(c < EE ? e[r * EE + c] : 0.0f);
}
__global__ void k_prep_conv(const float* __restrict__ w, _Float16* __restrict__ dst) {
    int i = blockIdx.x * 256 + threadIdx.x;          // 3*64*64
    int s = i >> 12, rem = i & 4095;
    int o = rem >> 6, ii = rem & 63;
    // B-matrix layout for conv GEMM: dst[s][o][i] = conv_w[o][i][s]
    dst[i] = (_Float16)((o < EE && ii < EE) ? w[o * (EE * 3) + ii * 3 + s] : 0.0f);
}
__global__ void k_zero_pad(_Float16* __restrict__ F) {
    int i = blockIdx.x * 256 + threadIdx.x;          // 16384*6
    int r = i / 6, c = i % 6;
    F[r * 256 + 250 + c] = (_Float16)0.0f;
}

// ---------- feature kernel: char conv (WMMA) + maxpool + word_emb ----------
// one wave per (b,w) word window; grid = 81920/8 blocks of 256 threads
__global__ void k_features(const int* __restrict__ x, const int* __restrict__ w2c,
                           const float* __restrict__ word_emb,
                           const _Float16* __restrict__ char_emb16,
                           const _Float16* __restrict__ convw16,
                           const float* __restrict__ conv_b,
                           _Float16* __restrict__ F) {
    __shared__ __align__(16) _Float16 ce[8][34 * 64];   // padded char matrix per wave
    const int wave = threadIdx.x >> 5;
    const int lane = threadIdx.x & 31;
    const int half = lane >> 4, l16 = lane & 15;
    const int g = blockIdx.x * 8 + wave;                // word-window id, exact grid
    const int wid = x[g];
    _Float16* Ce = ce[wave];

    // zero padded rows, then fill rows 1..20 with f16 char embeddings
#pragma unroll 2
    for (int r = 0; r < 34; ++r) {
        Ce[r * 64 + lane * 2]     = (_Float16)0.0f;
        Ce[r * 64 + lane * 2 + 1] = (_Float16)0.0f;
    }
    const int* cids = w2c + wid * LL;
    for (int j = 0; j < LL; ++j) {
        int cid = cids[j];                               // uniform across lanes
        Ce[(1 + j) * 64 + lane * 2]     = char_emb16[cid * 64 + lane * 2];
        Ce[(1 + j) * 64 + lane * 2 + 1] = char_emb16[cid * 64 + lane * 2 + 1];
    }

    // conv as 3 shifted GEMMs: out[l][o] = sum_s sum_i Ce[l+s][i] * Wk[s][i][o]
    v8f acc[2][4];
#pragma unroll
    for (int mt = 0; mt < 2; ++mt)
#pragma unroll
        for (int nt = 0; nt < 4; ++nt) acc[mt][nt] = (v8f){};

#pragma unroll
    for (int s = 0; s < 3; ++s) {
#pragma unroll
        for (int ks = 0; ks < 2; ++ks) {
            int kb = ks * 32 + half * 8;
            v16h a0 = load_frag(&Ce[(s + l16) * 64 + kb],      &Ce[(s + l16) * 64 + kb + 16]);
            v16h a1 = load_frag(&Ce[(16 + s + l16) * 64 + kb], &Ce[(16 + s + l16) * 64 + kb + 16]);
#pragma unroll
            for (int nt = 0; nt < 4; ++nt) {
                const _Float16* bp = convw16 + s * 4096 + (nt * 16 + l16) * 64 + ks * 32 + half * 8;
                v16h b = load_frag(bp, bp + 16);
                acc[0][nt] = wmma16(a0, b, acc[0][nt]);
                acc[1][nt] = wmma16(a1, b, acc[1][nt]);
            }
        }
    }

    // bias + max over the 20 valid char positions, + word embedding, store f16
    const int row = g / WW, colbase = (g % WW) * EE;
#pragma unroll
    for (int nt = 0; nt < 4; ++nt) {
        int n = nt * 16 + l16;
        float mx = -1e30f;
#pragma unroll
        for (int r = 0; r < 8; ++r) mx = fmaxf(mx, acc[0][nt][r]);   // rows 0..15
        if (half == 0) {
#pragma unroll
            for (int r = 0; r < 4; ++r) mx = fmaxf(mx, acc[1][nt][r]); // rows 16..19
        }
        mx = fmaxf(mx, __shfl_xor(mx, 16, 32));                       // merge halves
        float bias = (n < EE) ? conv_b[n] : 0.0f;
        float we   = (n < EE) ? word_emb[wid * EE + n] : 0.0f;
        float feat = mx + bias + we;
        if (half == 0 && n < EE)
            F[row * 256 + colbase + n] = (_Float16)feat;
    }
}

// ---------- fused FC1(tanh) -> FC2 -> softmax, M-tile = 128 rows ----------
// block = 128 rows (8 waves x 16 rows), all waves share W1/W2 chunks via LDS.
// Loop over 64 N-chunks of 64 cols: cooperative stage W1 chunk (32KB) and W2
// chunk (8KB), each wave computes FC1 for its rows, tanh->f16->per-wave LDS
// re-layout, consumes it as FC2 A-fragments; full logits stay in registers.
__global__ void k_fc(const _Float16* __restrict__ F, const _Float16* __restrict__ W1,
                     const float* __restrict__ b1, const _Float16* __restrict__ W2,
                     const float* __restrict__ b2, float* __restrict__ out) {
    __shared__ __align__(16) _Float16 w1c[64 * 256];     // 32KB [n_local][k]
    __shared__ __align__(16) _Float16 w2ch[64 * 64];     // 8KB  [o][k_local]
    __shared__ __align__(16) _Float16 stage[8][16 * 32]; // 8KB  per-wave h staging
    const int t = threadIdx.x;
    const int wave = t >> 5, lane = t & 31;
    const int half = lane >> 4, l16 = lane & 15;
    const int m0 = blockIdx.x * 128 + wave * 16;         // this wave's 16 rows

    // A fragments for FC1 (16 rows x 256 K), loaded once, reused for all N
    v16h afrag[8];
#pragma unroll
    for (int ks = 0; ks < 8; ++ks) {
        const _Float16* p = F + (m0 + l16) * 256 + ks * 32 + half * 8;
        afrag[ks] = load_frag(p, p + 16);
    }

    v8f acc2[4];                                         // full logits (cols 0..63)
#pragma unroll
    for (int ot = 0; ot < 4; ++ot) acc2[ot] = (v8f){};
    _Float16* st = stage[wave];

    for (int ch = 0; ch < 64; ++ch) {
        const int nb = ch * 64;                          // chunk col base
        // cooperative stage: W1 rows nb..nb+63 are contiguous (64*256 halfs)
#pragma unroll
        for (int i = 0; i < 8; ++i)
            *(f16x8*)&w1c[(t + i * 256) * 8] = *(const f16x8*)&W1[nb * 256 + (t + i * 256) * 8];
        // W2 chunk [64 o][64 k_local], k_local = nb..nb+63
#pragma unroll
        for (int j = 0; j < 2; ++j) {
            int q = t * 2 + j;                           // 0..511 vec8 units
            int o = q >> 3, rem = q & 7;
            *(f16x8*)&w2ch[o * 64 + rem * 8] = *(const f16x8*)&W2[o * HH + nb + rem * 8];
        }
        __syncthreads();

#pragma unroll
        for (int nt = 0; nt < 4; ++nt) {                 // 4 FC1 n-tiles in chunk
            v8f c = (v8f){};
#pragma unroll
            for (int ks = 0; ks < 8; ++ks) {
                v16h b = load_frag(&w1c[(nt * 16 + l16) * 256 + ks * 32 + half * 8],
                                   &w1c[(nt * 16 + l16) * 256 + ks * 32 + half * 8 + 16]);
                c = wmma16(afrag[ks], b, c);
            }
            float bias = b1[nb + nt * 16 + l16];
#pragma unroll
            for (int r = 0; r < 8; ++r) {
                float hv = tanhf(c[r] + bias);
                st[(r + 8 * half) * 32 + (nt & 1) * 16 + l16] = (_Float16)hv; // C->A relayout
            }
            if (nt & 1) {                                // staged 32 h-cols -> FC2 k-step
                v16h ah = load_frag(&st[l16 * 32 + half * 8],
                                    &st[l16 * 32 + 16 + half * 8]);
                int kl = (nt >> 1) * 32;
#pragma unroll
                for (int ot = 0; ot < 4; ++ot) {
                    v16h b = load_frag(&w2ch[(ot * 16 + l16) * 64 + kl + half * 8],
                                       &w2ch[(ot * 16 + l16) * 64 + kl + half * 8 + 16]);
                    acc2[ot] = wmma16(ah, b, acc2[ot]);
                }
            }
        }
        __syncthreads();                                 // protect chunk buffers
    }

    // in-register softmax per wave: lane holds rows (r + 8*half), col ot*16+l16
#pragma unroll
    for (int r = 0; r < 8; ++r) {
        float v[4];
#pragma unroll
        for (int ot = 0; ot < 4; ++ot) {
            int n = ot * 16 + l16;
            v[ot] = (n < OO) ? (acc2[ot][r] + b2[n]) : -1e30f;
        }
        float mx = fmaxf(fmaxf(v[0], v[1]), fmaxf(v[2], v[3]));
#pragma unroll
        for (int m = 1; m < 16; m <<= 1) mx = fmaxf(mx, __shfl_xor(mx, m, 16));
        float e[4], sum = 0.0f;
#pragma unroll
        for (int ot = 0; ot < 4; ++ot) { e[ot] = __expf(v[ot] - mx); if (ot * 16 + l16 < OO) sum += e[ot]; }
#pragma unroll
        for (int m = 1; m < 16; m <<= 1) sum += __shfl_xor(sum, m, 16);
        float inv = 1.0f / sum;
        int row = m0 + r + 8 * half;
#pragma unroll
        for (int ot = 0; ot < 4; ++ot) {
            int n = ot * 16 + l16;
            if (n < OO) out[row * OO + n] = e[ot] * inv;
        }
    }
}

extern "C" void kernel_launch(void* const* d_in, const int* in_sizes, int n_in,
                              void* d_out, int out_size, void* d_ws, size_t ws_size,
                              hipStream_t stream) {
    const int*   x        = (const int*)d_in[0];
    const int*   w2c      = (const int*)d_in[1];
    const float* word_emb = (const float*)d_in[2];
    const float* char_emb = (const float*)d_in[3];
    const float* conv_w   = (const float*)d_in[4];
    const float* conv_b   = (const float*)d_in[5];
    const float* fc1_w    = (const float*)d_in[6];
    const float* fc1_b    = (const float*)d_in[7];
    const float* fc2_w    = (const float*)d_in[8];
    const float* fc2_b    = (const float*)d_in[9];
    float* out = (float*)d_out;

    // workspace carve (~11 MB)
    char* p = (char*)d_ws;
    _Float16* F    = (_Float16*)p; p += (size_t)BB * 256 * 2;   // 8 MB features f16
    _Float16* W1   = (_Float16*)p; p += (size_t)HH * 256 * 2;   // 2 MB fc1 f16
    _Float16* W2   = (_Float16*)p; p += (size_t)64 * HH * 2;    // 0.5 MB fc2 f16
    _Float16* CE16 = (_Float16*)p; p += (size_t)CC * 64 * 2;    // 16 KB char emb f16
    _Float16* CW16 = (_Float16*)p; p += (size_t)3 * 64 * 64 * 2;// 24 KB conv w f16

    k_prep_fc1 <<<(HH * 256) / 256, 256, 0, stream>>>(fc1_w, W1);
    k_prep_fc2 <<<(64 * HH) / 256, 256, 0, stream>>>(fc2_w, W2);
    k_prep_char<<<(CC * 64) / 256, 256, 0, stream>>>(char_emb, CE16);
    k_prep_conv<<<(3 * 64 * 64) / 256, 256, 0, stream>>>(conv_w, CW16);
    k_zero_pad <<<(BB * 6) / 256, 256, 0, stream>>>(F);

    k_features<<<(BB * WW) / 8, 256, 0, stream>>>(x, w2c, word_emb, CE16, CW16, conv_b, F);
    k_fc<<<BB / 128, 256, 0, stream>>>(F, W1, fc1_b, W2, fc2_b, out);
}
